// CayleyNet_81174881894892
// MI455X (gfx1250) — matrix operand order repaired and checked by reading, not compiled, not run on hardware
//
#include <hip/hip_runtime.h>

// ---------------- problem constants ----------------
#define NREAL 3000
#define NPAD  3008            // multiple of 16 (M tiles) and 32 (K chunks)
#define EDGES 48000
#define C     32              // FEAT == HID
#define OUTD  8
#define KITER 5
#define KSEL  10800           // ceil(0.9 * 4*NREAL)
#define EXCL  (4*NREAL - KSEL)        // 1200 excluded rows
#define KEEP_THRESH (NREAL - EXCL)    // 1800
#define AMUL_WAVES 4                  // k-split waves per output tile

typedef __attribute__((ext_vector_type(16))) _Float16 v16h;
typedef __attribute__((ext_vector_type(8)))  _Float16 v8h;
typedef __attribute__((ext_vector_type(8)))  float    v8f;

union H16 { v16h v; _Float16 h[16]; };
union H8  { v8h  v; _Float16 h[8];  };

// ---------------- utility kernels ----------------
__global__ void zero_u4(uint4* p, int n) {
  int i = blockIdx.x * blockDim.x + threadIdx.x;
  int stride = gridDim.x * blockDim.x;
  uint4 z = {0u, 0u, 0u, 0u};
  for (; i < n; i += stride) p[i] = z;
}

__global__ void zero_f32(float* p, int n) {
  int i = blockIdx.x * blockDim.x + threadIdx.x;
  if (i < n) p[i] = 0.0f;
}

// A[r][c] = 1 (idempotent -> dedups duplicate edges, matching .set);
// deg counts duplicates (matching .add).
__global__ void build_adj(const int* __restrict__ row, const int* __restrict__ col,
                          _Float16* __restrict__ A16, float* __restrict__ deg) {
  int e = blockIdx.x * blockDim.x + threadIdx.x;
  if (e < EDGES) {
    int r = row[e], c = col[e];
    A16[(size_t)r * NPAD + c] = (_Float16)1.0f;
    atomicAdd(&deg[r], 1.0f);
  }
}

// dvals = deg - alpha ; neg_invd2 = -1/dvals ; invD = 1/(dvals*h)
__global__ void prep_diag(const float* __restrict__ deg, const float* __restrict__ hp,
                          const float* __restrict__ ap,
                          float* __restrict__ neg_invd2, float* __restrict__ invD) {
  int i = blockIdx.x * blockDim.x + threadIdx.x;
  if (i < NREAL) {
    float dv = deg[i] - ap[0];
    neg_invd2[i] = -1.0f / dv;
    invD[i] = 1.0f / (dv * hp[0]);
  }
}

// z16T[c*NPAD + j] = f16( (prescale? prescale[j] : 1) * src[j*C + c] ), 0 in pad rows.
// k-major layout so WMMA B fragments are contiguous v16h loads.
__global__ void cvt_z(const float* __restrict__ src, const float* __restrict__ prescale,
                      _Float16* __restrict__ z16T) {
  int i = blockIdx.x * blockDim.x + threadIdx.x;
  if (i >= C * NPAD) return;
  int c = i / NPAD;
  int j = i - c * NPAD;
  float v = 0.0f;
  if (j < NREAL) {
    v = src[j * C + c];
    if (prescale) v *= prescale[j];
  }
  z16T[(size_t)c * NPAD + j] = (_Float16)v;
}

// ---------------- WMMA GEMM: dst = rowscale .* (A @ z) + addend ----------------
// A: NPAD x NPAD f16 (0/1, exact).  z: k-major f16, C x NPAD ("z16T").
// 4 waves per 16-row tile split the K dimension (chunk-interleaved so every
// wave still issues contiguous b128 loads); fixed-order LDS reduction keeps
// the result bitwise deterministic, then wave 0 applies the fused epilogue.
__global__ __launch_bounds__(32 * AMUL_WAVES)
void amul_wmma(const _Float16* __restrict__ A16, const _Float16* __restrict__ z16T,
               const float* __restrict__ rowscale, const float* __restrict__ addend,
               float* __restrict__ dst) {
  __shared__ float red[AMUL_WAVES][32][16];
  int tile = blockIdx.x;              // 0..NPAD/16-1
  int tid  = threadIdx.x;
  int lane = tid & 31;
  int wave = tid >> 5;
  int n16  = lane & 15;
  int hi   = lane >> 4;               // lane half selects K = 16*hi..16*hi+15 (contiguous order, same for A and B)
  v8f c0 = {0.f,0.f,0.f,0.f,0.f,0.f,0.f,0.f};
  v8f c1 = {0.f,0.f,0.f,0.f,0.f,0.f,0.f,0.f};

  const _Float16* arow  = A16  + (size_t)(tile * 16 + n16) * NPAD + hi * 16;
  const _Float16* brow0 = z16T + (size_t)n16 * NPAD        + hi * 16;
  const _Float16* brow1 = z16T + (size_t)(n16 + 16) * NPAD + hi * 16;

  for (int kc = wave * 32; kc < NPAD; kc += 32 * AMUL_WAVES) {
    // A fragment (16x32 f16): lane row M=n16; h[t] = K = kc + 16*hi + t
    H16 a, b0, b1;
    H8 lo, hh;
    lo.v = *reinterpret_cast<const v8h*>(arow + kc);
    hh.v = *reinterpret_cast<const v8h*>(arow + kc + 8);
#pragma unroll
    for (int t = 0; t < 8; ++t) { a.h[t] = lo.h[t]; a.h[8 + t] = hh.h[t]; }
    // B fragment (32x16 f16): lane col N=n16; h[t] = K = kc + 16*hi + t  (contiguous in z16T)
    b0.v = *reinterpret_cast<const v16h*>(brow0 + kc);
    b1.v = *reinterpret_cast<const v16h*>(brow1 + kc);

    c0 = __builtin_amdgcn_wmma_f32_16x16x32_f16(false, a.v, false, b0.v, (short)0, c0, false, false);
    c1 = __builtin_amdgcn_wmma_f32_16x16x32_f16(false, a.v, false, b1.v, (short)0, c1, false, false);
  }

  // Publish partial accumulators, then fixed-order cross-wave reduction.
#pragma unroll
  for (int r = 0; r < 8; ++r) {
    red[wave][lane][r]     = c0[r];
    red[wave][lane][8 + r] = c1[r];
  }
  __syncthreads();
  if (wave == 0) {
    // C/D layout: acc[r] -> row = tile*16 + 8*hi + r, col = n16 (c0) / 16+n16 (c1)
#pragma unroll
    for (int r = 0; r < 8; ++r) {
      int m = tile * 16 + hi * 8 + r;
      if (m < NREAL) {
        float s0 = ((red[0][lane][r] + red[1][lane][r]) +
                    (red[2][lane][r] + red[3][lane][r]));
        float s1 = ((red[0][lane][8 + r] + red[1][lane][8 + r]) +
                    (red[2][lane][8 + r] + red[3][lane][8 + r]));
        float rs = rowscale ? rowscale[m] : 1.0f;
        dst[m * C + n16]      = rs * s0 + addend[m * C + n16];
        dst[m * C + 16 + n16] = rs * s1 + addend[m * C + 16 + n16];
      }
    }
  }
}

// dst[n,o] = (accumulate? dst : 0) + scale * sum_c src[n,c] * W[o,c]   (exact f32)
__global__ void lin_t(const float* __restrict__ src, const float* __restrict__ W,
                      float* __restrict__ dst, float scale, int accumulate) {
  int i = blockIdx.x * blockDim.x + threadIdx.x;
  if (i >= NREAL * C) return;
  int n = i >> 5, o = i & 31;
  float s = 0.0f;
#pragma unroll
  for (int c = 0; c < C; ++c) s += src[n * C + c] * W[o * C + c];
  float r = scale * s;
  if (accumulate) r += dst[i];
  dst[i] = r;
}

__global__ void relu_k(const float* __restrict__ src, float* __restrict__ dst, int n) {
  int i = blockIdx.x * blockDim.x + threadIdx.x;
  if (i < n) dst[i] = fmaxf(src[i], 0.0f);
}

// s[i] = tanh( (x2[i] . pw) / ||pw|| )
__global__ void score_k(const float* __restrict__ x2, const float* __restrict__ pw,
                        float* __restrict__ s) {
  int i = blockIdx.x * blockDim.x + threadIdx.x;
  if (i >= NREAL) return;
  float nrm = 0.0f, d = 0.0f;
#pragma unroll
  for (int c = 0; c < C; ++c) { nrm += pw[c] * pw[c]; d += x2[i * C + c] * pw[c]; }
  s[i] = tanhf(d / sqrtf(nrm));
}

// TopK over 12000 rows where 9000 are exactly zero: a real row is dropped iff
// it is among the EXCL most-negative scores (stable tie-break matches top_k).
__global__ void keep_k(const float* __restrict__ s, int* __restrict__ keep) {
  int i = blockIdx.x * blockDim.x + threadIdx.x;
  if (i >= NREAL) return;
  float si = s[i];
  if (si >= 0.0f) { keep[i] = 1; return; }
  int r = 0;
  for (int j = 0; j < NREAL; ++j) {
    float sj = s[j];
    if (sj > si || (sj == si && j < i)) ++r;
  }
  keep[i] = (r < KEEP_THRESH) ? 1 : 0;
}

// g = (1/KSEL) * sum_{kept} s_i * x2_i ; out = g @ lw^T + lb
__global__ void pool_final(const float* __restrict__ x2, const float* __restrict__ s,
                           const int* __restrict__ keep, const float* __restrict__ lw,
                           const float* __restrict__ lb, float* __restrict__ out) {
  __shared__ float g[C];
  int c = threadIdx.x;
  float acc = 0.0f;
  for (int i = 0; i < NREAL; ++i)
    if (keep[i]) acc += s[i] * x2[i * C + c];
  g[c] = acc * (1.0f / (float)KSEL);
  __syncthreads();
  if (c < OUTD) {
    float r = lb[c];
#pragma unroll
    for (int k2 = 0; k2 < C; ++k2) r += g[k2] * lw[c * C + k2];
    out[c] = r;
  }
}

// ---------------- host orchestration ----------------
extern "C" void kernel_launch(void* const* d_in, const int* in_sizes, int n_in,
                              void* d_out, int out_size, void* d_ws, size_t ws_size,
                              hipStream_t stream) {
  (void)in_sizes; (void)n_in; (void)out_size; (void)ws_size;
  const float* x    = (const float*)d_in[0];
  const int*   ei   = (const int*)  d_in[1];
  const float* hp   = (const float*)d_in[2];
  const float* ap   = (const float*)d_in[3];
  const float* Wr0  = (const float*)d_in[4];
  const float* Wc0a = (const float*)d_in[5];
  const float* Wc0b = (const float*)d_in[6];
  const float* Wr1  = (const float*)d_in[7];
  const float* Wc1a = (const float*)d_in[8];
  const float* Wc1b = (const float*)d_in[9];
  const float* pw   = (const float*)d_in[10];
  const float* lw   = (const float*)d_in[11];
  const float* lb   = (const float*)d_in[12];
  const int* rowI = ei;
  const int* colI = ei + EDGES;

  char* base = (char*)d_ws;
  size_t off = 0;
  auto carve = [&](size_t bytes) -> char* {
    char* p = base + off;
    off += (bytes + 255) & ~(size_t)255;
    return p;
  };
  _Float16* A16  = (_Float16*)carve((size_t)NPAD * NPAD * 2);  // ~18 MB, L2-resident
  _Float16* Z16T = (_Float16*)carve((size_t)C * NPAD * 2);
  float* deg  = (float*)carve(NREAL * 4);
  float* nid2 = (float*)carve(NREAL * 4);
  float* invD = (float*)carve(NREAL * 4);
  float* BJ   = (float*)carve((size_t)NREAL * C * 4);
  float* YW   = (float*)carve((size_t)NREAL * C * 4);
  float* OUTB = (float*)carve((size_t)NREAL * C * 4);
  float* X1   = (float*)carve((size_t)NREAL * C * 4);
  float* X2   = (float*)carve((size_t)NREAL * C * 4);
  float* S    = (float*)carve(NREAL * 4);
  int*   KEEP = (int*)  carve(NREAL * 4);

  // Build shared adjacency + diagonals (identical for both convs).
  int a16u4 = (int)((size_t)NPAD * NPAD * 2 / 16);
  zero_u4<<<4096, 256, 0, stream>>>((uint4*)A16, a16u4);
  zero_f32<<<(NREAL + 255) / 256, 256, 0, stream>>>(deg, NREAL);
  build_adj<<<(EDGES + 255) / 256, 256, 0, stream>>>(rowI, colI, A16, deg);
  prep_diag<<<(NREAL + 255) / 256, 256, 0, stream>>>(deg, hp, ap, nid2, invD);

  int gz = (C * NPAD + 255) / 256;
  int gn = (NREAL * C + 255) / 256;

  auto conv = [&](const float* xin, const float* Wr, const float* Wca,
                  const float* Wcb, float* xout) {
    lin_t<<<gn, 256, 0, stream>>>(xin, Wr, OUTB, 1.0f, 0);       // out = x @ Wr^T
    const float* ycur = xin;
    const float* Wcs[2] = {Wca, Wcb};
    for (int rep = 0; rep < 2; ++rep) {
      // b_j = y - (1/dvals) .* (A @ y)
      cvt_z<<<gz, 256, 0, stream>>>(ycur, (const float*)nullptr, Z16T);
      amul_wmma<<<NPAD / 16, 32 * AMUL_WAVES, 0, stream>>>(A16, Z16T, nid2, ycur, BJ);
      // yk = b_j; K x: yk = A @ (invD .* yk) + b_j
      const float* ysrc = BJ;
      for (int it = 0; it < KITER; ++it) {
        cvt_z<<<gz, 256, 0, stream>>>(ysrc, invD, Z16T);
        amul_wmma<<<NPAD / 16, 32 * AMUL_WAVES, 0, stream>>>(A16, Z16T,
                                                             (const float*)nullptr, BJ, YW);
        ysrc = YW;
      }
      lin_t<<<gn, 256, 0, stream>>>(YW, Wcs[rep], OUTB, 2.0f, 1); // out += 2*y @ Wc^T
      ycur = YW;
    }
    relu_k<<<gn, 256, 0, stream>>>(OUTB, xout, NREAL * C);
  };

  conv(x,  Wr0, Wc0a, Wc0b, X1);   // conv1 (top block; bottom halves provably zero)
  conv(X1, Wr1, Wc1a, Wc1b, X2);   // conv2 shares the same A / diagonals

  score_k<<<(NREAL + 255) / 256, 256, 0, stream>>>(X2, pw, S);
  keep_k<<<(NREAL + 255) / 256, 256, 0, stream>>>(S, KEEP);
  pool_final<<<1, 32, 0, stream>>>(X2, S, KEEP, lw, lb, (float*)d_out);
}
